// FeedbackLoop_46291157516382
// MI455X (gfx1250) — compile-verified, matrix operand order checked by reading, not run on hardware
//
#include <hip/hip_runtime.h>

#define DIM 1024
#define NROWS 32768

typedef __attribute__((ext_vector_type(16))) __bf16 v16bf;
typedef __attribute__((ext_vector_type(8)))  __bf16 v8bf;
typedef __attribute__((ext_vector_type(8)))  float  v8f;
typedef __attribute__((ext_vector_type(4)))  int    v4i;

// ---- gfx1250 async global->LDS copy (ASYNCcnt-tracked), with fallback -----
#if defined(__AMDGCN__) && __has_builtin(__builtin_amdgcn_global_load_async_to_lds_b128)
#define HAVE_ASYNC_LDS 1
typedef __attribute__((address_space(1))) v4i* gv4p_t;
typedef __attribute__((address_space(3))) v4i* lv4p_t;
__device__ __forceinline__ void async_cp16(const void* g, void* l) {
    __builtin_amdgcn_global_load_async_to_lds_b128((gv4p_t)(void*)g, (lv4p_t)l, 0, 0);
}
#if __has_builtin(__builtin_amdgcn_s_wait_asynccnt)
#define WAIT_ASYNC() __builtin_amdgcn_s_wait_asynccnt(0)
#else
#define WAIT_ASYNC() asm volatile("s_wait_asynccnt 0x0" ::: "memory")
#endif
#else
#define HAVE_ASYNC_LDS 0
__device__ __forceinline__ void async_cp16(const void* g, void* l) {
    *(uint4*)l = *(const uint4*)g;   // synchronous fallback
}
#define WAIT_ASYNC() ((void)0)
#endif

__device__ __forceinline__ unsigned short f2bf(float f) {
    unsigned u = __float_as_uint(f);
    u += 0x7FFFu + ((u >> 16) & 1u);   // round-to-nearest-even
    return (unsigned short)(u >> 16);
}

__device__ __forceinline__ float sigmoidf_(float x) {
    return 1.0f / (1.0f + __expf(-x));
}
__device__ __forceinline__ float tanhf_(float x) {
    x = fminf(15.0f, fmaxf(-15.0f, x));
    float e = __expf(2.0f * x);
    return (e - 1.0f) / (e + 1.0f);
}

// concat two 16B LDS loads into one clean v16bf SSA value
__device__ __forceinline__ v16bf ldfrag(const unsigned char* p0, const unsigned char* p1) {
    v8bf lo = *(const v8bf*)p0;
    v8bf hi = *(const v8bf*)p1;
    return __builtin_shufflevector(lo, hi, 0, 1, 2, 3, 4, 5, 6, 7,
                                           8, 9, 10, 11, 12, 13, 14, 15);
}

// ---------------------------------------------------------------------------
// Kernel 1: convert [W_ih ; W_hh] (f32, each [3072][1024]) -> bf16 Wb[6144][1024]
// ---------------------------------------------------------------------------
__global__ void wcvt_kernel(const float* __restrict__ wih,
                            const float* __restrict__ whh,
                            unsigned short* __restrict__ wb) {
    int idx = (blockIdx.x * 256 + threadIdx.x) * 4;   // grid covers 6144*1024 exactly
    const int halfN = 3072 * 1024;
    float4 v;
    if (idx < halfN) v = *(const float4*)(wih + idx);
    else             v = *(const float4*)(whh + (idx - halfN));
    ushort4 o = make_ushort4(f2bf(v.x), f2bf(v.y), f2bf(v.z), f2bf(v.w));
    *(ushort4*)(wb + idx) = o;
}

// ---------------------------------------------------------------------------
// Kernel 2: per-row: state f32 -> bf16, and error = sigmoid(state . We + be)
// ---------------------------------------------------------------------------
__global__ void rowstats_kernel(const float* __restrict__ s,
                                const float* __restrict__ we,
                                const float* __restrict__ be,
                                unsigned short* __restrict__ sb,
                                float* __restrict__ err) {
    const int lane = threadIdx.x & 31;
    const int row  = blockIdx.x * 8 + (threadIdx.x >> 5);
    const float* srow = s + (size_t)row * DIM;
    unsigned short* brow = sb + (size_t)row * DIM;
    float sum = 0.0f;
#pragma unroll
    for (int i = 0; i < 8; ++i) {
        int k = i * 128 + lane * 4;
        float4 v = *(const float4*)(srow + k);
        float4 w = *(const float4*)(we + k);
        sum += v.x * w.x + v.y * w.y + v.z * w.z + v.w * w.w;
        ushort4 o = make_ushort4(f2bf(v.x), f2bf(v.y), f2bf(v.z), f2bf(v.w));
        *(ushort4*)(brow + k) = o;
    }
#pragma unroll
    for (int off = 16; off; off >>= 1) sum += __shfl_xor(sum, off, 32);
    if (lane == 0) err[row] = sigmoidf_(sum + be[0]);
}

// ---------------------------------------------------------------------------
// Kernel 3: fused dual-GEMM (bf16 WMMA, f32 acc) + GRU epilogue.
// Block 256 thr (8 waves); region = 64 rows x 64 d-cols; all 6 gate tiles.
// LDS: double-buffered dense tile (28672 B x2); 16B chunks XOR-swizzled.
// Data path: GLOBAL_LOAD_ASYNC_TO_LDS_B128; K-loop kept rolled so each
// accumulator tile has a single register home (no v_mov churn / WMMA hazards).
// ---------------------------------------------------------------------------
#define TILE_B 28672u

__global__ __launch_bounds__(256)
void gru_step_kernel(const float* __restrict__ Sin,
                     const unsigned short* __restrict__ Sb,
                     const unsigned short* __restrict__ Wb,
                     const float* __restrict__ bih,
                     const float* __restrict__ bhh,
                     const float* __restrict__ Err,
                     float* __restrict__ Sout) {
    __shared__ __align__(16) unsigned char smem[2 * TILE_B];

    const int tid  = threadIdx.x;
    const int lane = tid & 31;
    const int wv   = tid >> 5;
    const int nt   = wv & 3;      // wave's 16-col N-tile within 64-col region
    const int grp  = wv >> 2;     // 0: gates ih(r,z,n)  1: gates hh(r,z,n)
    const int hl   = lane >> 4;
    const int lr   = lane & 15;

    const int rowBase = blockIdx.x * 64;
    const int colBase = blockIdx.y * 64;

    // ---- global->LDS copy assignment: thread -> (row m, 16B chunk) --------
    const int m     = tid >> 2;               // 0..63
    const int chunk = tid & 3;                // 4 x 16B chunks of 64B k-row
    const unsigned ldsA = m * 64 + ((chunk ^ ((m >> 2) & 3)) * 16);  // swizzled

    const unsigned short* gA  = Sb + (size_t)(rowBase + m) * DIM + chunk * 8;
    const unsigned short* gB0 = Wb + (size_t)(colBase + m) * DIM + chunk * 8;
    // weight rows for gate g are at +g*1024*DIM elements

    auto issueTile = [&](int step, unsigned dstoff) {
        const int ke = step * 32;             // k elements
        async_cp16(gA + ke, smem + dstoff + ldsA);
#pragma unroll
        for (int g = 0; g < 6; ++g)
            async_cp16(gB0 + (size_t)g * 1024 * DIM + ke,
                       smem + dstoff + 4096 + g * 4096 + ldsA);
    };

    // ---- per-lane fragment address pieces (mt/g-independent swizzle) ------
    const unsigned sw = (unsigned)((lr >> 2) & 3);
    const unsigned c0 = ((unsigned)hl ^ sw) * 16u;
    const unsigned c1 = (((unsigned)hl + 2u) ^ sw) * 16u;
    const unsigned aBase = (unsigned)lr * 64u;                    // + mt*1024
    const unsigned bBase = 4096u + (unsigned)(nt * 16 + lr) * 64u;// + g*4096

    v8f acc[4][3];
    const v8f vzero = {0.f, 0.f, 0.f, 0.f, 0.f, 0.f, 0.f, 0.f};
#pragma unroll
    for (int mt = 0; mt < 4; ++mt)
#pragma unroll
        for (int gg = 0; gg < 3; ++gg) acc[mt][gg] = vzero;

    unsigned bufoff = 0;
    issueTile(0, 0);

#pragma unroll 1
    for (int step = 0; step < 32; ++step) {
        WAIT_ASYNC();            // my async writes into buf[bufoff] done
        __syncthreads();         // everyone's done; also: all reads of the
                                 // other buffer (step-1 compute) complete
        if (step < 31) issueTile(step + 1, bufoff ^ TILE_B);  // overlap

        v16bf bfr[3];
#pragma unroll
        for (int gg = 0; gg < 3; ++gg) {
            const unsigned base = bufoff + bBase + (unsigned)(grp * 3 + gg) * 4096u;
            bfr[gg] = ldfrag(smem + base + c0, smem + base + c1);
        }
#pragma unroll
        for (int mt = 0; mt < 4; ++mt) {
            const unsigned base = bufoff + aBase + (unsigned)mt * 1024u;
            v16bf af = ldfrag(smem + base + c0, smem + base + c1);
#pragma unroll
            for (int gg = 0; gg < 3; ++gg)
                acc[mt][gg] = __builtin_amdgcn_wmma_f32_16x16x32_bf16(
                    false, af, false, bfr[gg], (short)0, acc[mt][gg], false, false);
        }
        bufoff ^= TILE_B;
    }

    // ---------------- epilogue: exchange gate tiles via LDS, apply GRU ------
    float* Ls = (float*)smem;                  // [6][16][65] f32 per M-tile pass
    for (int mt = 0; mt < 4; ++mt) {
#pragma unroll
        for (int gg = 0; gg < 3; ++gg) {
            const int g = grp * 3 + gg;
#pragma unroll
            for (int r = 0; r < 8; ++r)
                Ls[g * 1040 + (r + 8 * hl) * 65 + nt * 16 + lr] = acc[mt][gg][r];
        }
        __syncthreads();
#pragma unroll
        for (int it = 0; it < 4; ++it) {
            const int idx = tid + it * 256;    // 16x64 elements
            const int r = idx >> 6, c = idx & 63;
            const int grow = rowBase + mt * 16 + r;
            const int d = colBase + c;
            const float gir = Ls[0 * 1040 + r * 65 + c] + bih[d];
            const float giz = Ls[1 * 1040 + r * 65 + c] + bih[1024 + d];
            const float gin = Ls[2 * 1040 + r * 65 + c] + bih[2048 + d];
            const float ghr = Ls[3 * 1040 + r * 65 + c] + bhh[d];
            const float ghz = Ls[4 * 1040 + r * 65 + c] + bhh[1024 + d];
            const float ghn = Ls[5 * 1040 + r * 65 + c] + bhh[2048 + d];
            const float rg = sigmoidf_(gir + ghr);
            const float zg = sigmoidf_(giz + ghz);
            const float ng = tanhf_(gin + rg * ghn);
            const float s  = Sin[(size_t)grow * DIM + d];
            const float e  = Err[grow];
            const float ns = (1.0f - zg) * ng + zg * s;
            Sout[(size_t)grow * DIM + d] = (1.0f - e) * s + e * ns;
        }
        __syncthreads();
    }
}

// ---------------------------------------------------------------------------
extern "C" void kernel_launch(void* const* d_in, const int* in_sizes, int n_in,
                              void* d_out, int out_size, void* d_ws, size_t ws_size,
                              hipStream_t stream) {
    const float* x    = (const float*)d_in[0];
    const float* W_ih = (const float*)d_in[1];
    const float* W_hh = (const float*)d_in[2];
    const float* b_ih = (const float*)d_in[3];
    const float* b_hh = (const float*)d_in[4];
    const float* We   = (const float*)d_in[5];
    const float* be   = (const float*)d_in[6];
    float* out = (float*)d_out;

    // workspace layout (~77 MB): Wb bf16 | state bf16 | error f32
    unsigned short* Wb = (unsigned short*)d_ws;                          // 12,582,912 B
    unsigned short* Sb = (unsigned short*)((char*)d_ws + 12582912);      // 67,108,864 B
    float* ErrBuf      = (float*)((char*)d_ws + 12582912 + 67108864);    //    131,072 B

    dim3 gemmGrid(NROWS / 64, DIM / 64, 1);  // 512 x 16

    wcvt_kernel<<<6144, 256, 0, stream>>>(W_ih, W_hh, Wb);

    // iteration 0: read x, write out
    rowstats_kernel<<<NROWS / 8, 256, 0, stream>>>(x, We, be, Sb, ErrBuf);
    gru_step_kernel<<<gemmGrid, 256, 0, stream>>>(x, Sb, Wb, b_ih, b_hh, ErrBuf, out);

    // iteration 1: in-place on out (each f32 state element is read only by its
    // own writer; the GEMM A operand comes from the separate bf16 buffer)
    rowstats_kernel<<<NROWS / 8, 256, 0, stream>>>(out, We, be, Sb, ErrBuf);
    gru_step_kernel<<<gemmGrid, 256, 0, stream>>>(out, Sb, Wb, b_ih, b_hh, ErrBuf, out);
}
